// GCRNN_42984032698745
// MI455X (gfx1250) — matrix operand order repaired and checked by reading, not compiled, not run on hardware
//
#include <hip/hip_runtime.h>
#include <hip/hip_bf16.h>

// ---------- types ----------
typedef __bf16 bf16;
typedef bf16         bf16x16 __attribute__((ext_vector_type(16)));
typedef float        f32x8   __attribute__((ext_vector_type(8)));
typedef unsigned int u32x8   __attribute__((ext_vector_type(8)));

__device__ __forceinline__ unsigned short f2bf(float f) {
    unsigned int u = __builtin_bit_cast(unsigned int, f);
    u += 0x7FFFu + ((u >> 16) & 1u);          // round-to-nearest-even
    return (unsigned short)(u >> 16);
}

__device__ __forceinline__ float sigmoidf(float x) {
    return 1.0f / (1.0f + expf(-x));
}

#define GEMM_ACC  1
#define GEMM_RELU 2

// ---------------------------------------------------------------------------
// WMMA bf16 GEMM: C[M x F] (+)= A[M x K] @ W[K x F]
//   A  : bf16, row-major [M,K]
//   Wp : bf16, pre-packed per-lane B fragments: Wp[((kt*F + n) << 5) + kk],
//        kt = k/32, kk = k%32  ->  each lane fetches 2 x b128.
//   C  : f32 (ldc stride), optional accumulate / bias / relu epilogue.
// Block: 256 threads = 8 waves; tile = 32 rows x 128 cols.
//   Each wave owns a 32x16 C tile (2 accumulators) sharing one B fragment,
//   so each K-step issues 2 WMMAs per B fetch.
// Grid:  (M/32, F/128).  Requires M%32==0, K/32 even, F%128==0.
// ---------------------------------------------------------------------------
__device__ __forceinline__ void stage_tile(unsigned short* __restrict__ dst,
                                           const unsigned short* __restrict__ Ag,
                                           long rowbase, int k0, int arow, int aq) {
    const uint2 v = *(const uint2*)(Ag + rowbase + k0 + aq * 4);
    *(uint2*)&dst[arow * 36 + aq * 4] = v;
}

__device__ __forceinline__ bf16x16 load_afrag(const unsigned short* __restrict__ sbuf, int abase) {
    u32x8 au;
#pragma unroll
    for (int v = 0; v < 8; ++v) {
        const int kb = 2 * v;
        au[v] = *(const unsigned int*)&sbuf[abase + kb];
    }
    return __builtin_bit_cast(bf16x16, au);
}

__device__ __forceinline__ bf16x16 load_bfrag(const unsigned short* __restrict__ bptr) {
    const uint4* bp = (const uint4*)bptr;     // 32B-aligned fragment
    uint4 b0 = bp[0], b1 = bp[1];
    u32x8 bu;
    bu[0] = b0.x; bu[1] = b0.y; bu[2] = b0.z; bu[3] = b0.w;
    bu[4] = b1.x; bu[5] = b1.y; bu[6] = b1.z; bu[7] = b1.w;
    return __builtin_bit_cast(bf16x16, bu);
}

__global__ void __launch_bounds__(256)
wmma_gemm_kernel(const unsigned short* __restrict__ A, const unsigned short* __restrict__ Wp,
                 float* __restrict__ C, const float* __restrict__ bias,
                 int K, int F, int ldc, int flags)
{
    __shared__ __align__(16) unsigned short sA[2][32 * 36];

    const int tid  = threadIdx.x;
    const int lane = tid & 31;
    const int wave = tid >> 5;         // 0..7 -> column subtile
    const int row0 = blockIdx.x * 32;
    const int col0 = blockIdx.y * 128 + wave * 16;
    const bool hi  = lane >= 16;
    const int  l16 = lane & 15;
    const int  n   = col0 + l16;

    const int arow = tid >> 3;   // 0..31 : row of A tile this thread stages
    const int aq   = tid & 7;    // 4-half slot within the 32-wide K chunk
    const long rowbase = (long)(row0 + arow) * K;
    const int  abase0  = l16 * 36        + (hi ? 16 : 0);   // rows 0..15
    const int  abase1  = (16 + l16) * 36 + (hi ? 16 : 0);   // rows 16..31

    const float bb = bias ? bias[n] : 0.0f;           // hoisted epilogue bias

    f32x8 acc0 = {}, acc1 = {};
    if (flags & GEMM_ACC) {
#pragma unroll
        for (int r = 0; r < 8; ++r) {
            int m = row0 + r + (hi ? 8 : 0);
            acc0[r] = C[(long)m * ldc + n];
            acc1[r] = C[(long)(m + 16) * ldc + n];
        }
    }

    const int  nkt   = K >> 5;                         // always even (2,4,8)
    const long bstep = (long)F << 5;                   // fragment stride per kt
    long bofs        = ((long)n << 5) + (hi ? 16 : 0); // fragment base for kt=0

    stage_tile(sA[0], A, rowbase, 0, arow, aq);
    for (int kt = 0; kt < nkt; kt += 2) {
        __syncthreads();
        stage_tile(sA[1], A, rowbase, (kt + 1) * 32, arow, aq);
        {
            bf16x16 b  = load_bfrag(Wp + bofs);
            bf16x16 a0 = load_afrag(sA[0], abase0);
            bf16x16 a1 = load_afrag(sA[0], abase1);
            acc0 = __builtin_amdgcn_wmma_f32_16x16x32_bf16(false, a0, false, b, (short)0, acc0, false, false);
            acc1 = __builtin_amdgcn_wmma_f32_16x16x32_bf16(false, a1, false, b, (short)0, acc1, false, false);
        }
        bofs += bstep;
        __syncthreads();
        if (kt + 2 < nkt) stage_tile(sA[0], A, rowbase, (kt + 2) * 32, arow, aq);
        {
            bf16x16 b  = load_bfrag(Wp + bofs);
            bf16x16 a0 = load_afrag(sA[1], abase0);
            bf16x16 a1 = load_afrag(sA[1], abase1);
            acc0 = __builtin_amdgcn_wmma_f32_16x16x32_bf16(false, a0, false, b, (short)0, acc0, false, false);
            acc1 = __builtin_amdgcn_wmma_f32_16x16x32_bf16(false, a1, false, b, (short)0, acc1, false, false);
        }
        bofs += bstep;
    }

#pragma unroll
    for (int r = 0; r < 8; ++r) {
        const int m = row0 + r + (hi ? 8 : 0);
        float v0 = acc0[r] + bb;
        float v1 = acc1[r] + bb;
        if (flags & GEMM_RELU) { v0 = fmaxf(v0, 0.0f); v1 = fmaxf(v1, 0.0f); }
        C[(long)m * ldc + n]        = v0;
        C[(long)(m + 16) * ldc + n] = v1;
    }
}

// ---------------------------------------------------------------------------
// Pre-pack / convert kernels
// ---------------------------------------------------------------------------
// W[K,F] f32 -> Wp fragment layout bf16
__global__ void pack_w_kernel(const float* __restrict__ W, unsigned short* __restrict__ Wp,
                              int K, int F) {
    int i = blockIdx.x * blockDim.x + threadIdx.x;
    if (i >= K * F) return;
    int k = i / F, nn = i - k * F;
    int kt = k >> 5, kk = k & 31;
    Wp[(((long)kt * F + nn) << 5) + kk] = f2bf(W[i]);
}

// f32 -> bf16, 4 elements per thread
__global__ void cvt4_bf16_kernel(const float4* __restrict__ in, uint2* __restrict__ out,
                                 int total4) {
    int i = blockIdx.x * blockDim.x + threadIdx.x;
    if (i >= total4) return;
    float4 v = in[i];
    uint2 p;
    p.x = (unsigned int)f2bf(v.x) | ((unsigned int)f2bf(v.y) << 16);
    p.y = (unsigned int)f2bf(v.z) | ((unsigned int)f2bf(v.w) << 16);
    out[i] = p;
}

// ---------------------------------------------------------------------------
// Graph / elementwise kernels
// ---------------------------------------------------------------------------
__global__ void fill_kernel(float* p, int n, float v) {
    int i = blockIdx.x * blockDim.x + threadIdx.x;
    if (i < n) p[i] = v;
}

__global__ void deg_edges_kernel(const long long* __restrict__ ei,
                                 float* __restrict__ deg, int E) {
    int e = blockIdx.x * blockDim.x + threadIdx.x;
    if (e < E) atomicAdd(&deg[(int)ei[(long)E + e]], 1.0f);
}

__global__ void rsqrt_kernel(float* p, int n) {
    int i = blockIdx.x * blockDim.x + threadIdx.x;
    if (i < n) p[i] = rsqrtf(p[i]);
}

// one thread per (edge, 4-float chunk); F = 4 << qshift
__global__ void scatter_kernel(const long long* __restrict__ ei,
                               const float* __restrict__ dinv,
                               const float* __restrict__ xw,
                               float* __restrict__ agg, int E, int qshift) {
    int i = blockIdx.x * blockDim.x + threadIdx.x;
    int total = E << qshift;
    if (i >= total) return;
    int e = i >> qshift;
    int q = i & ((1 << qshift) - 1);
    int s = (int)ei[e];
    int d = (int)ei[(long)E + e];
    float norm = dinv[s] * dinv[d];
    const float4 v = *(const float4*)(xw + ((long)s << (qshift + 2)) + (q << 2));
    float* ap = agg + ((long)d << (qshift + 2)) + (q << 2);
    atomicAdd(ap + 0, v.x * norm);
    atomicAdd(ap + 1, v.y * norm);
    atomicAdd(ap + 2, v.z * norm);
    atomicAdd(ap + 3, v.w * norm);
}

// h = relu(agg + xw*dinv^2 + bias); f32 in-place  (F = 1 << fshift)
__global__ void selfloop_bias_relu_kernel(float* __restrict__ h,
                                          const float* __restrict__ xw,
                                          const float* __restrict__ dinv,
                                          const float* __restrict__ bias,
                                          int total, int fshift) {
    int i = blockIdx.x * blockDim.x + threadIdx.x;
    if (i >= total) return;
    int f  = i & ((1 << fshift) - 1);
    int nn = i >> fshift;
    float dv = dinv[nn];
    h[i] = fmaxf(h[i] + xw[i] * dv * dv + bias[f], 0.0f);
}

// same epilogue but emits bf16 (consumer is a GEMM A operand only)
__global__ void selfloop_bias_relu_bf16_kernel(const float* __restrict__ agg,
                                               const float* __restrict__ xw,
                                               const float* __restrict__ dinv,
                                               const float* __restrict__ bias,
                                               unsigned short* __restrict__ out,
                                               int total, int fshift) {
    int i = blockIdx.x * blockDim.x + threadIdx.x;
    if (i >= total) return;
    int f  = i & ((1 << fshift) - 1);
    int nn = i >> fshift;
    float dv = dinv[nn];
    out[i] = f2bf(fmaxf(agg[i] + xw[i] * dv * dv + bias[f], 0.0f));
}

// xin_bf[n, 0:128] = z[n], xin_bf[n, 128:256] = x[n]   (bf16 output)
__global__ void concat_bf16_kernel(unsigned short* __restrict__ xin,
                                   const float* __restrict__ z,
                                   const float* __restrict__ x, int total) {
    int i = blockIdx.x * blockDim.x + threadIdx.x;
    if (i >= total) return;
    int f  = i & 255;
    int nn = i >> 8;
    float v = (f < 128) ? z[((long)nn << 7) + f] : x[((long)nn << 7) + (f - 128)];
    xin[i] = f2bf(v);
}

// rh_bf = bf16(prev_h * sigmoid(Pr))
__global__ void gru_rh_bf16_kernel(unsigned short* __restrict__ out,
                                   const float* __restrict__ Pr,
                                   const float* __restrict__ prev_h, int total) {
    int i = blockIdx.x * blockDim.x + threadIdx.x;
    if (i >= total) return;
    out[i] = f2bf(prev_h[i] * sigmoidf(Pr[i]));
}

// Hn = sigmoid(Pz)*prev_h + (1-sigmoid(Pz))*tanh(Hn)   (Hn holds Ph on entry)
__global__ void gru_final_kernel(float* __restrict__ Hn,
                                 const float* __restrict__ Pz,
                                 const float* __restrict__ prev_h, int total) {
    int i = blockIdx.x * blockDim.x + threadIdx.x;
    if (i >= total) return;
    float zv = sigmoidf(Pz[i]);
    float ht = tanhf(Hn[i]);
    Hn[i] = zv * prev_h[i] + (1.0f - zv) * ht;
}

// mean-pool z per graph (batch sorted) -> fused[b, 0:128]; no atomics
__device__ __forceinline__ int lower_bound_ll(const long long* a, int n, long long v) {
    int lo = 0, hi = n;
    while (lo < hi) { int mid = (lo + hi) >> 1; if (a[mid] < v) lo = mid + 1; else hi = mid; }
    return lo;
}

__global__ void __launch_bounds__(256)
pool_kernel(const long long* __restrict__ batch, const float* __restrict__ z,
            float* __restrict__ fused, int n) {
    __shared__ int   sr[2];
    __shared__ float part[128];
    const int b = blockIdx.x;
    if (threadIdx.x == 0) {
        sr[0] = lower_bound_ll(batch, n, (long long)b);
        sr[1] = lower_bound_ll(batch, n, (long long)b + 1);
    }
    __syncthreads();
    const int lo = sr[0], hiN = sr[1];
    const int f    = threadIdx.x & 127;
    const int half = threadIdx.x >> 7;
    float acc = 0.0f;
    for (int i = lo + half; i < hiN; i += 2) acc += z[((long)i << 7) + f];
    if (half) part[f] = acc;
    __syncthreads();
    if (!half) {
        float s   = acc + part[f];
        float cnt = (float)(hiN - lo);
        fused[b * 256 + f] = s / fmaxf(cnt, 1.0f);
    }
}

// hh = relu(fused@Wh1+bh1); out = hh@Wh2+bh2   (one block per graph)
__global__ void __launch_bounds__(128)
head_kernel(const float* __restrict__ fused, const float* __restrict__ Wh1,
            const float* __restrict__ bh1, const float* __restrict__ Wh2,
            const float* __restrict__ bh2, float* __restrict__ out) {
    __shared__ float sf[256];
    __shared__ float sh[128];
    const int b = blockIdx.x, t = threadIdx.x;
    sf[t]       = fused[b * 256 + t];
    sf[t + 128] = fused[b * 256 + 128 + t];
    __syncthreads();
    float acc = bh1[t];
#pragma unroll 4
    for (int k = 0; k < 256; ++k) acc += sf[k] * Wh1[k * 128 + t];
    sh[t] = fmaxf(acc, 0.0f) * Wh2[t];
    __syncthreads();
    for (int s = 64; s > 0; s >>= 1) {
        if (t < s) sh[t] += sh[t + s];
        __syncthreads();
    }
    if (t == 0) out[b] = sh[0] + bh2[0];
}

// ---------------------------------------------------------------------------
// launch
// ---------------------------------------------------------------------------
static inline int ceil_div(long a, int b) { return (int)((a + b - 1) / b); }

extern "C" void kernel_launch(void* const* d_in, const int* in_sizes, int n_in,
                              void* d_out, int out_size, void* d_ws, size_t ws_size,
                              hipStream_t stream) {
    const float*     x      = (const float*)d_in[0];
    const long long* ei     = (const long long*)d_in[1];
    const float*     u      = (const float*)d_in[2];
    const long long* batch  = (const long long*)d_in[3];
    const float*     prev_h = (const float*)d_in[5];
    const float *W1 = (const float*)d_in[6],  *b1 = (const float*)d_in[7];
    const float *W2 = (const float*)d_in[8],  *b2 = (const float*)d_in[9];
    const float *Wg = (const float*)d_in[10], *bg = (const float*)d_in[11];
    const float *Wxz = (const float*)d_in[12], *Whz = (const float*)d_in[13], *bz = (const float*)d_in[14];
    const float *Wxr = (const float*)d_in[15], *Whr = (const float*)d_in[16], *br = (const float*)d_in[17];
    const float *Wxh = (const float*)d_in[18], *Whh = (const float*)d_in[19], *bh = (const float*)d_in[20];
    const float *Wh1 = (const float*)d_in[21], *bh1 = (const float*)d_in[22];
    const float *Wh2 = (const float*)d_in[23], *bh2 = (const float*)d_in[24];

    const int N = in_sizes[0] / 128;   // 100000
    const int E = in_sizes[1] / 2;     // 800000
    const int B = in_sizes[2] / 64;    // 64

    // ---- workspace carve-out (256B aligned chunks) ------------------------
    char* wp = (char*)d_ws;
    auto alloc = [&](size_t bytes) -> char* {
        char* r = wp;
        wp += (bytes + 255) & ~(size_t)255;
        return r;
    };
    float* deg  = (float*)alloc((size_t)N * 4);              // degree -> dinv
    float* bufA = (float*)alloc((size_t)N * 256 * 4);        // xw1 (f32)
    float* bufB = (float*)alloc((size_t)N * 256 * 4);        // agg1 -> Pz
    float* bufC = (float*)alloc((size_t)N * 128 * 4);        // xw2 ; later aliased by rh_bf
    float* bufD = (float*)alloc((size_t)N * 128 * 4);        // agg2 / z
    float* bufE = (float*)alloc((size_t)N * 256 * 4);        // Pr
    unsigned short* bf1 = (unsigned short*)alloc((size_t)N * 256 * 2); // x_bf -> xin_bf
    unsigned short* bf2 = (unsigned short*)alloc((size_t)N * 256 * 2); // h1_bf -> prevh_bf
    unsigned short* bf3 = (unsigned short*)bufC;             // rh_bf (lifetime-disjoint alias)
    float* fused = (float*)alloc((size_t)B * 256 * 4);
    unsigned short* ubf  = (unsigned short*)alloc(64 * 64 * 2);
    unsigned short* W1p  = (unsigned short*)alloc((size_t)128 * 256 * 2);
    unsigned short* W2p  = (unsigned short*)alloc((size_t)256 * 128 * 2);
    unsigned short* Wxzp = (unsigned short*)alloc((size_t)256 * 256 * 2);
    unsigned short* Whzp = (unsigned short*)alloc((size_t)256 * 256 * 2);
    unsigned short* Wxrp = (unsigned short*)alloc((size_t)256 * 256 * 2);
    unsigned short* Whrp = (unsigned short*)alloc((size_t)256 * 256 * 2);
    unsigned short* Wxhp = (unsigned short*)alloc((size_t)256 * 256 * 2);
    unsigned short* Whhp = (unsigned short*)alloc((size_t)256 * 256 * 2);
    unsigned short* Wgp  = (unsigned short*)alloc((size_t)64 * 128 * 2);

    float* outp = (float*)d_out;
    float* Hn   = outp + (long)B;              // [N,256] (holds Ph first)

    const int T = 256;
    auto packW = [&](const float* W, unsigned short* Wp, int K, int F) {
        pack_w_kernel<<<ceil_div((long)K * F, T), T, 0, stream>>>(W, Wp, K, F);
    };

    // ---- weight pre-pack (bf16 WMMA fragment layout) ----------------------
    packW(W1, W1p, 128, 256);
    packW(W2, W2p, 256, 128);
    packW(Wxz, Wxzp, 256, 256); packW(Whz, Whzp, 256, 256);
    packW(Wxr, Wxrp, 256, 256); packW(Whr, Whrp, 256, 256);
    packW(Wxh, Wxhp, 256, 256); packW(Whh, Whhp, 256, 256);
    packW(Wg, Wgp, 64, 128);

    // ---- symmetric normalization ------------------------------------------
    fill_kernel<<<ceil_div(N, T), T, 0, stream>>>(deg, N, 1.0f);
    deg_edges_kernel<<<ceil_div(E, T), T, 0, stream>>>(ei, deg, E);
    rsqrt_kernel<<<ceil_div(N, T), T, 0, stream>>>(deg, N);

    // ---- GCN layer 1: h1 = relu(agg(x@W1) + b1) ---------------------------
    cvt4_bf16_kernel<<<ceil_div((long)N * 32, T), T, 0, stream>>>((const float4*)x, (uint2*)bf1, N * 32);
    wmma_gemm_kernel<<<dim3(N / 32, 2), T, 0, stream>>>(bf1, W1p, bufA, nullptr, 128, 256, 256, 0);
    fill_kernel<<<ceil_div((long)N * 256, T), T, 0, stream>>>(bufB, N * 256, 0.0f);
    scatter_kernel<<<ceil_div((long)E << 6, T), T, 0, stream>>>(ei, deg, bufA, bufB, E, 6);
    selfloop_bias_relu_bf16_kernel<<<ceil_div((long)N * 256, T), T, 0, stream>>>(bufB, bufA, deg, b1, bf2, N * 256, 8);

    // ---- GCN layer 2: z = relu(agg(h1@W2) + b2) ---------------------------
    wmma_gemm_kernel<<<dim3(N / 32, 1), T, 0, stream>>>(bf2, W2p, bufC, nullptr, 256, 128, 128, 0);
    fill_kernel<<<ceil_div((long)N * 128, T), T, 0, stream>>>(bufD, N * 128, 0.0f);
    scatter_kernel<<<ceil_div((long)E << 5, T), T, 0, stream>>>(ei, deg, bufC, bufD, E, 5);
    selfloop_bias_relu_kernel<<<ceil_div((long)N * 128, T), T, 0, stream>>>(bufD, bufC, deg, b2, N * 128, 7);

    // ---- GConvGRU ---------------------------------------------------------
    concat_bf16_kernel<<<ceil_div((long)N * 256, T), T, 0, stream>>>(bf1, bufD, x, N * 256);   // xin_bf
    cvt4_bf16_kernel<<<ceil_div((long)N * 64, T), T, 0, stream>>>((const float4*)prev_h, (uint2*)bf2, N * 64);

    // Pz = xin@Wxz + prev_h@Whz + bz
    wmma_gemm_kernel<<<dim3(N / 32, 2), T, 0, stream>>>(bf1, Wxzp, bufB, nullptr, 256, 256, 256, 0);
    wmma_gemm_kernel<<<dim3(N / 32, 2), T, 0, stream>>>(bf2, Whzp, bufB, bz, 256, 256, 256, GEMM_ACC);
    // Pr = xin@Wxr + prev_h@Whr + br
    wmma_gemm_kernel<<<dim3(N / 32, 2), T, 0, stream>>>(bf1, Wxrp, bufE, nullptr, 256, 256, 256, 0);
    wmma_gemm_kernel<<<dim3(N / 32, 2), T, 0, stream>>>(bf2, Whrp, bufE, br, 256, 256, 256, GEMM_ACC);
    // rh = prev_h * sigmoid(Pr)  (bf16, aliases dead xw2 buffer)
    gru_rh_bf16_kernel<<<ceil_div((long)N * 256, T), T, 0, stream>>>(bf3, bufE, prev_h, N * 256);
    // Ph = xin@Wxh + rh@Whh + bh   (into Hn region of d_out)
    wmma_gemm_kernel<<<dim3(N / 32, 2), T, 0, stream>>>(bf1, Wxhp, Hn, nullptr, 256, 256, 256, 0);
    wmma_gemm_kernel<<<dim3(N / 32, 2), T, 0, stream>>>(bf3, Whhp, Hn, bh, 256, 256, 256, GEMM_ACC);
    // Hn = Z*prev_h + (1-Z)*tanh(Ph)
    gru_final_kernel<<<ceil_div((long)N * 256, T), T, 0, stream>>>(Hn, bufB, prev_h, N * 256);

    // ---- pooling + global projection + head -------------------------------
    pool_kernel<<<B, 256, 0, stream>>>(batch, bufD, fused, N);                   // fused[:, :128]
    cvt4_bf16_kernel<<<ceil_div(64 * 16, T), T, 0, stream>>>((const float4*)u, (uint2*)ubf, 64 * 16);
    wmma_gemm_kernel<<<dim3(B / 32, 1), T, 0, stream>>>(ubf, Wgp, fused + 128, bg, 64, 128, 256, GEMM_RELU);
    head_kernel<<<B, 128, 0, stream>>>(fused, Wh1, bh1, Wh2, bh2, outp);
}
